// MistralAttentionForSVDKVTrain_76519137346146
// MI455X (gfx1250) — matrix operand order repaired and checked by reading, not compiled, not run on hardware
//
#include <hip/hip_runtime.h>

// ---------------------------------------------------------------------------
// MI455X (gfx1250, wave32) implementation of MistralAttentionForSVDKVTrain.
// All GEMMs + attention use v_wmma_f32_16x16x32_bf16 (fp32->bf16 conversion
// via native v_cvt_pk_bf16_f32 during LDS staging; fp32 accumulate).
// ~330 GFLOP total, ~340 MB of HBM traffic -> memory-bound on 23.3 TB/s HBM.
// Staging is software-pipelined: global loads issue before the workgroup
// barrier so they overlap the previous tile's WMMA work.
// ---------------------------------------------------------------------------

typedef __attribute__((ext_vector_type(16))) __bf16 v16bf;
typedef __attribute__((ext_vector_type(2)))  __bf16 v2bf;
typedef __attribute__((ext_vector_type(8)))  float  v8f;

struct U128 { unsigned int a, b, c, d; };
union Frag { v16bf v; U128 q[2]; };

// Native bf16 converts (lower to v_cvt_pk_bf16_f32 for pairs).
__device__ __forceinline__ unsigned short f2bf(float f) {
  __bf16 h = (__bf16)f;
  return __builtin_bit_cast(unsigned short, h);
}
__device__ __forceinline__ unsigned int pack2bf(float a, float b) {
  v2bf v = { (__bf16)a, (__bf16)b };
  return __builtin_bit_cast(unsigned int, v);
}

__device__ __forceinline__ v8f wmma_bf16(v16bf a, v16bf b, v8f c) {
  return __builtin_amdgcn_wmma_f32_16x16x32_bf16(
      /*neg_a=*/false, a, /*neg_b=*/false, b,
      /*c_mod=*/(short)0, c, /*reuse_a=*/false, /*reuse_b=*/false);
}

// constants of the problem
#define SEQ  2048
#define HID  4096
#define NHQ  32
#define NKVH 8
#define HDIM 128
#define KCD  64   // compressed dims per kv head (KC == VC == 64)

// ---------------------------------------------------------------------------
// Generic C = A(MxK,f32) * B(KxN,f32) GEMM, bf16 WMMA, fp32 accum.
// Block = 256 threads (8 waves), tile 128x128, K-step 32.
// ---------------------------------------------------------------------------
__launch_bounds__(256)
__global__ void gemm_bf16_wmma(const float* __restrict__ A,
                               const float* __restrict__ B,
                               float* __restrict__ C,
                               int M, int N, int K) {
  __shared__ unsigned short As[128][40];   // [m][k], 80B rows (16B aligned)
  __shared__ unsigned short Bs[128][40];   // transposed: [n][k]
  const int t    = threadIdx.x;
  const int m0   = blockIdx.y * 128;
  const int n0   = blockIdx.x * 128;
  const int wv   = t >> 5, lane = t & 31;
  const int wm   = wv >> 2, wn = wv & 3;    // 2x4 wave grid over 128x128
  const int hl   = lane >> 4, l15 = lane & 15;

  const v8f vz = {0.f,0.f,0.f,0.f,0.f,0.f,0.f,0.f};
  v8f acc[4][2];
  #pragma unroll
  for (int mi = 0; mi < 4; ++mi)
    #pragma unroll
    for (int ni = 0; ni < 2; ++ni) acc[mi][ni] = vz;

  const int arow = t >> 3, akq = (t & 7) << 2;   // A staging: 8 thr/row
  const int bk   = t >> 5, bn4 = (t & 31) << 2;  // B staging

  for (int kk = 0; kk < K; kk += 32) {
    // Issue this tile's global loads into registers BEFORE the barrier so
    // they overlap the previous tile's WMMA work in the other waves.
    float4 fa[4], fb[4];
    #pragma unroll
    for (int i = 0; i < 4; ++i)
      fa[i] = *(const float4*)(A + (size_t)(m0 + arow + (i << 5)) * K + kk + akq);
    #pragma unroll
    for (int i = 0; i < 4; ++i)
      fb[i] = *(const float4*)(B + (size_t)(kk + bk + (i << 3)) * N + n0 + bn4);
    if (kk + 32 < K) {                           // global_prefetch_b8
      __builtin_prefetch(A + (size_t)(m0 + arow) * K + kk + 32 + akq);
      __builtin_prefetch(B + (size_t)(kk + 32 + bk) * N + n0 + bn4);
    }

    __syncthreads();                             // LDS reuse fence
    #pragma unroll
    for (int i = 0; i < 4; ++i) {                // A tile 128x32 -> bf16 LDS
      unsigned int* dst = (unsigned int*)&As[arow + (i << 5)][akq];
      dst[0] = pack2bf(fa[i].x, fa[i].y);
      dst[1] = pack2bf(fa[i].z, fa[i].w);
    }
    #pragma unroll
    for (int i = 0; i < 4; ++i) {                // B tile 32x128 -> Bt[n][k]
      int k = bk + (i << 3);
      Bs[bn4+0][k] = f2bf(fb[i].x);  Bs[bn4+1][k] = f2bf(fb[i].y);
      Bs[bn4+2][k] = f2bf(fb[i].z);  Bs[bn4+3][k] = f2bf(fb[i].w);
    }
    __syncthreads();

    // A fragments: lane m=l15, K runs {k0..k0+7, k0+16..k0+23}, k0 = 8*hl
    Frag af[4];
    const int ak0 = hl ? 8 : 0;
    #pragma unroll
    for (int mi = 0; mi < 4; ++mi) {
      int r = wm * 64 + mi * 16 + l15;
      af[mi].q[0] = *(const U128*)&As[r][ak0];
      af[mi].q[1] = *(const U128*)&As[r][ak0 + 16];
    }
    // B fragments: lane n=l15, K runs {k0..k0+7, k0+16..k0+23}, k0 = 8*hl (from Bt[n][k])
    Frag bfr[2];
    const int bk0 = hl ? 8 : 0;
    #pragma unroll
    for (int ni = 0; ni < 2; ++ni) {
      int r = wn * 32 + ni * 16 + l15;
      bfr[ni].q[0] = *(const U128*)&Bs[r][bk0];
      bfr[ni].q[1] = *(const U128*)&Bs[r][bk0 + 16];
    }
    #pragma unroll
    for (int mi = 0; mi < 4; ++mi)
      #pragma unroll
      for (int ni = 0; ni < 2; ++ni)
        acc[mi][ni] = wmma_bf16(af[mi].v, bfr[ni].v, acc[mi][ni]);
  }

  // C layout: N = l15, M = 8*hl + vgpr
  #pragma unroll
  for (int mi = 0; mi < 4; ++mi)
    #pragma unroll
    for (int ni = 0; ni < 2; ++ni)
      #pragma unroll
      for (int v = 0; v < 8; ++v) {
        int row = m0 + wm * 64 + mi * 16 + hl * 8 + v;
        int col = n0 + wn * 32 + ni * 16 + l15;
        C[(size_t)row * N + col] = acc[mi][ni][v];
      }
}

// ---------------------------------------------------------------------------
// Flash attention (causal, GQA 4:1). Block = 256 thr (8 waves) owns a
// 128-row Q tile of one head; each wave owns 16 rows. Key blocks of 64.
// Q fragments live in registers for the whole kernel.
// ---------------------------------------------------------------------------
__launch_bounds__(256)
__global__ void flash_attn_wmma(const float* __restrict__ Q,
                                const float* __restrict__ Kg,
                                const float* __restrict__ Vg,
                                float* __restrict__ O) {
  __shared__ unsigned short Kt[64][128];   // [key][d]  (row-major == B frag)
  __shared__ unsigned short Vt[128][80];   // [d][key]  (transposed)
  __shared__ unsigned short Pt[128][64];   // [qrow][key] bf16 probabilities

  const int t    = threadIdx.x;
  const int wv   = t >> 5, lane = t & 31;
  const int hl   = lane >> 4, l15 = lane & 15;
  const int h    = blockIdx.y, kvh = h >> 2;
  const int q0   = blockIdx.x * 128;

  // Q fragments (4 K-chunks of 32 dims), straight from global in A-layout
  Frag qf[4];
  {
    const float* qb = Q + (size_t)(q0 + wv * 16 + l15) * HID + h * HDIM;
    #pragma unroll
    for (int c = 0; c < 4; ++c) {
      int d0 = c * 32 + (hl ? 8 : 0);
      float4 f0 = *(const float4*)(qb + d0);
      float4 f1 = *(const float4*)(qb + d0 + 4);
      float4 f2 = *(const float4*)(qb + d0 + 16);
      float4 f3 = *(const float4*)(qb + d0 + 20);
      qf[c].q[0].a = pack2bf(f0.x, f0.y);
      qf[c].q[0].b = pack2bf(f0.z, f0.w);
      qf[c].q[0].c = pack2bf(f1.x, f1.y);
      qf[c].q[0].d = pack2bf(f1.z, f1.w);
      qf[c].q[1].a = pack2bf(f2.x, f2.y);
      qf[c].q[1].b = pack2bf(f2.z, f2.w);
      qf[c].q[1].c = pack2bf(f3.x, f3.y);
      qf[c].q[1].d = pack2bf(f3.z, f3.w);
    }
  }

  const v8f vz = {0.f,0.f,0.f,0.f,0.f,0.f,0.f,0.f};
  v8f of[8];                     // 16 rows x 128 dims output accum
  #pragma unroll
  for (int dt = 0; dt < 8; ++dt) of[dt] = vz;
  float rmax[8], rsum[8];
  #pragma unroll
  for (int v = 0; v < 8; ++v) { rmax[v] = -3.0e38f; rsum[v] = 0.f; }

  for (int kb0 = 0; kb0 < q0 + 128; kb0 += 64) {
    // Stage K (row-major) + V (transposed) in two load-then-store batches so
    // the global loads clause up and overlap the barrier.
    #pragma unroll
    for (int half = 0; half < 2; ++half) {
      float4 fk[4], fv[4];
      #pragma unroll
      for (int i = 0; i < 4; ++i) {
        int idx = t + (((half << 2) + i) << 8);  // 0..2047
        int row = idx >> 5;                      // key 0..63
        int c4  = (idx & 31) << 2;               // dim 0..124
        const float* kp = Kg + (size_t)(kb0 + row) * (NKVH*HDIM) + kvh*HDIM + c4;
        const float* vp = Vg + (size_t)(kb0 + row) * (NKVH*HDIM) + kvh*HDIM + c4;
        fk[i] = *(const float4*)kp;
        fv[i] = *(const float4*)vp;
      }
      if (half == 0) __syncthreads();            // LDS reuse fence
      #pragma unroll
      for (int i = 0; i < 4; ++i) {
        int idx = t + (((half << 2) + i) << 8);
        int row = idx >> 5;
        int c4  = (idx & 31) << 2;
        unsigned int* dk = (unsigned int*)&Kt[row][c4];
        dk[0] = pack2bf(fk[i].x, fk[i].y);
        dk[1] = pack2bf(fk[i].z, fk[i].w);
        Vt[c4+0][row] = f2bf(fv[i].x);  Vt[c4+1][row] = f2bf(fv[i].y);
        Vt[c4+2][row] = f2bf(fv[i].z);  Vt[c4+3][row] = f2bf(fv[i].w);
      }
    }
    __syncthreads();

    // S = Q K^T : 16 rows x 64 keys for this wave
    v8f sfr[4];
    #pragma unroll
    for (int nt = 0; nt < 4; ++nt) sfr[nt] = vz;
    #pragma unroll
    for (int nt = 0; nt < 4; ++nt)
      #pragma unroll
      for (int c = 0; c < 4; ++c) {
        Frag kf;
        int r  = nt * 16 + l15;
        int k0 = c * 32 + (hl ? 8 : 0);
        kf.q[0] = *(const U128*)&Kt[r][k0];
        kf.q[1] = *(const U128*)&Kt[r][k0 + 16];
        sfr[nt] = wmma_bf16(qf[c].v, kf.v, sfr[nt]);
      }

    // scale + causal mask (element positions match C layout)
    const float scale = 0.08838834764831845f;   // 1/sqrt(128)
    #pragma unroll
    for (int nt = 0; nt < 4; ++nt)
      #pragma unroll
      for (int v = 0; v < 8; ++v) {
        int key = kb0 + nt * 16 + l15;
        int qr  = q0 + wv * 16 + hl * 8 + v;
        float val = sfr[nt][v] * scale;
        sfr[nt][v] = (key <= qr) ? val : -3.0e38f;
      }

    // online softmax; rows live across the 16 lanes of each half-wave
    #pragma unroll
    for (int v = 0; v < 8; ++v) {
      float mv = fmaxf(fmaxf(sfr[0][v], sfr[1][v]), fmaxf(sfr[2][v], sfr[3][v]));
      #pragma unroll
      for (int m = 1; m < 16; m <<= 1) mv = fmaxf(mv, __shfl_xor(mv, m, 32));
      float nm   = fmaxf(rmax[v], mv);
      float corr = __expf(rmax[v] - nm);
      rmax[v] = nm;
      float ps = 0.f;
      #pragma unroll
      for (int nt = 0; nt < 4; ++nt) {
        float p = __expf(sfr[nt][v] - nm);
        ps += p;
        Pt[wv * 16 + hl * 8 + v][nt * 16 + l15] = f2bf(p);  // same-wave LDS: in-order
      }
      #pragma unroll
      for (int m = 1; m < 16; m <<= 1) ps += __shfl_xor(ps, m, 32);
      rsum[v] = rsum[v] * corr + ps;
      #pragma unroll
      for (int dt = 0; dt < 8; ++dt) of[dt][v] *= corr;
    }

    // O += P V  (P from own LDS rows, V transposed = B fragments)
    #pragma unroll
    for (int c2 = 0; c2 < 2; ++c2) {
      Frag pf;
      int k0 = c2 * 32 + (hl ? 8 : 0);
      int pr = wv * 16 + l15;
      pf.q[0] = *(const U128*)&Pt[pr][k0];
      pf.q[1] = *(const U128*)&Pt[pr][k0 + 16];
      #pragma unroll
      for (int dt = 0; dt < 8; ++dt) {
        Frag vf;
        int vr  = dt * 16 + l15;
        int vk0 = c2 * 32 + (hl ? 8 : 0);
        vf.q[0] = *(const U128*)&Vt[vr][vk0];
        vf.q[1] = *(const U128*)&Vt[vr][vk0 + 16];
        of[dt] = wmma_bf16(pf.v, vf.v, of[dt]);
      }
    }
  }

  float rinv[8];
  #pragma unroll
  for (int v = 0; v < 8; ++v) rinv[v] = 1.0f / rsum[v];
  #pragma unroll
  for (int dt = 0; dt < 8; ++dt)
    #pragma unroll
    for (int v = 0; v < 8; ++v) {
      int row = q0 + wv * 16 + hl * 8 + v;
      int col = h * HDIM + dt * 16 + l15;
      O[(size_t)row * HID + col] = of[dt][v] * rinv[v];
    }
}

// ---------------------------------------------------------------------------
// Fake quant, group along channel (v_c): 64 contiguous floats per group.
// ---------------------------------------------------------------------------
__global__ void quant_group_kernel(float* __restrict__ x, int ngroups) {
  int g = blockIdx.x * blockDim.x + threadIdx.x;
  if (g >= ngroups) return;
  float* p = x + (size_t)g * 64;
  float mn = p[0], mx = p[0];
  for (int i = 1; i < 64; ++i) { float v = p[i]; mn = fminf(mn, v); mx = fmaxf(mx, v); }
  float sc = fmaxf((mx - mn) * (1.0f / 15.0f), 1e-8f);
  float inv = 1.0f / sc;
  for (int i = 0; i < 64; ++i) p[i] = rintf((p[i] - mn) * inv) * sc + mn;
}

// Fake quant per-channel over token windows (k_c): stride `cols` between tokens.
__global__ void quant_channel_kernel(float* __restrict__ x, int cols, int win, int ngroups) {
  int g = blockIdx.x * blockDim.x + threadIdx.x;
  if (g >= ngroups) return;
  int c = g % cols, w = g / cols;
  float* p = x + (size_t)w * win * cols + c;
  float mn = p[0], mx = p[0];
  for (int i = 1; i < win; ++i) { float v = p[(size_t)i * cols]; mn = fminf(mn, v); mx = fmaxf(mx, v); }
  float sc = fmaxf((mx - mn) * (1.0f / 15.0f), 1e-8f);
  float inv = 1.0f / sc;
  for (int i = 0; i < win; ++i) {
    float v = p[(size_t)i * cols];
    p[(size_t)i * cols] = rintf((v - mn) * inv) * sc + mn;
  }
}

// ---------------------------------------------------------------------------
// In-place RoPE on [SEQ, nh*128] buffer.
// ---------------------------------------------------------------------------
__global__ void rope_kernel(float* __restrict__ x, const int* __restrict__ pos, int nh) {
  int idx = blockIdx.x * blockDim.x + threadIdx.x;
  int total = SEQ * nh * 64;
  if (idx >= total) return;
  int d  = idx & 63;
  int h  = (idx >> 6) % nh;
  int si = idx / (64 * nh);
  float p   = (float)pos[si];
  float ang = p * __expf(-(float)d * 0.14391156516f);   // ln(10000)/64
  float sn, cs; __sincosf(ang, &sn, &cs);
  float* base = x + (size_t)si * nh * HDIM + h * HDIM;
  float x0 = base[d], x1 = base[d + 64];
  base[d]      = x0 * cs - x1 * sn;
  base[d + 64] = x1 * cs + x0 * sn;
}

// ---------------------------------------------------------------------------
// MSE reduction: atomicAdd of (a-b)^2 / n per wave.
// ---------------------------------------------------------------------------
__global__ void mse_kernel(const float* __restrict__ a, const float* __restrict__ b,
                           int n, float inv_n, float* __restrict__ out) {
  float s = 0.f;
  for (int i = blockIdx.x * blockDim.x + threadIdx.x; i < n; i += gridDim.x * blockDim.x) {
    float d = a[i] - b[i];
    s += d * d;
  }
  s *= inv_n;
  for (int m = 1; m < 32; m <<= 1) s += __shfl_xor(s, m, 32);
  if ((threadIdx.x & 31) == 0) atomicAdd(out, s);
}

__global__ void zero_tail_kernel(float* __restrict__ out) {
  if (threadIdx.x < 2) out[threadIdx.x] = 0.f;
}

// ---------------------------------------------------------------------------
extern "C" void kernel_launch(void* const* d_in, const int* in_sizes, int n_in,
                              void* d_out, int out_size, void* d_ws, size_t ws_size,
                              hipStream_t stream) {
  (void)in_sizes; (void)n_in; (void)out_size; (void)ws_size;
  const float* x   = (const float*)d_in[0];
  const int*   pos = (const int*)d_in[1];
  const float* Wq  = (const float*)d_in[2];
  const float* Wk  = (const float*)d_in[3];
  const float* Wv  = (const float*)d_in[4];
  const float* Wo  = (const float*)d_in[5];
  const float* Wka = (const float*)d_in[6];
  const float* Wkb = (const float*)d_in[7];
  const float* Wva = (const float*)d_in[8];
  const float* Wvb = (const float*)d_in[9];
  float* out = (float*)d_out;

  // workspace layout (fp32), ~104 MB total
  float* ws  = (float*)d_ws;
  float* q   = ws;  ws += (size_t)SEQ * HID;            // 2048x4096
  float* k_o = ws;  ws += (size_t)SEQ * NKVH * HDIM;    // 2048x1024
  float* v_o = ws;  ws += (size_t)SEQ * NKVH * HDIM;
  float* k_c = ws;  ws += (size_t)SEQ * NKVH * KCD;     // 2048x512
  float* v_c = ws;  ws += (size_t)SEQ * NKVH * KCD;
  float* k_r = ws;  ws += (size_t)SEQ * NKVH * HDIM;
  float* v_r = ws;  ws += (size_t)SEQ * NKVH * HDIM;
  float* att = ws;  ws += (size_t)SEQ * HID;

  const dim3 blk(256);
  // projections
  gemm_bf16_wmma<<<dim3(HID/128, SEQ/128), blk, 0, stream>>>(x, Wq,  q,   SEQ, HID,  HID);
  gemm_bf16_wmma<<<dim3(1024/128, SEQ/128), blk, 0, stream>>>(x, Wk,  k_o, SEQ, 1024, HID);
  gemm_bf16_wmma<<<dim3(1024/128, SEQ/128), blk, 0, stream>>>(x, Wv,  v_o, SEQ, 1024, HID);
  gemm_bf16_wmma<<<dim3(512/128,  SEQ/128), blk, 0, stream>>>(x, Wka, k_c, SEQ, 512,  HID);
  gemm_bf16_wmma<<<dim3(512/128,  SEQ/128), blk, 0, stream>>>(x, Wva, v_c, SEQ, 512,  HID);

  // fake quant (in place; STE forward value == dequantized value)
  quant_channel_kernel<<<(32768 + 255)/256, blk, 0, stream>>>(k_c, NKVH*KCD, 32, 32768);
  quant_group_kernel<<<(16384 + 255)/256, blk, 0, stream>>>(v_c, SEQ * NKVH);

  // reconstruction up-projections
  gemm_bf16_wmma<<<dim3(1024/128, SEQ/128), blk, 0, stream>>>(k_c, Wkb, k_r, SEQ, 1024, 512);
  gemm_bf16_wmma<<<dim3(1024/128, SEQ/128), blk, 0, stream>>>(v_c, Wvb, v_r, SEQ, 1024, 512);

  // RoPE on q, k_o, k_r (not v)
  rope_kernel<<<(SEQ*NHQ*64 + 255)/256, blk, 0, stream>>>(q,   pos, NHQ);
  rope_kernel<<<(SEQ*NKVH*64 + 255)/256, blk, 0, stream>>>(k_o, pos, NKVH);
  rope_kernel<<<(SEQ*NKVH*64 + 255)/256, blk, 0, stream>>>(k_r, pos, NKVH);

  // distillation MSE losses -> out tail
  const int nmse = SEQ * NKVH * HDIM;       // 2,097,152
  zero_tail_kernel<<<1, 32, 0, stream>>>(out + (size_t)SEQ * HID);
  mse_kernel<<<512, blk, 0, stream>>>(k_o, k_r, nmse, 1.0f/(float)nmse, out + (size_t)SEQ*HID);
  mse_kernel<<<512, blk, 0, stream>>>(v_o, v_r, nmse, 1.0f/(float)nmse, out + (size_t)SEQ*HID + 1);

  // causal GQA flash attention on original K/V
  flash_attn_wmma<<<dim3(SEQ/128, NHQ), blk, 0, stream>>>(q, k_o, v_o, att);

  // output projection straight into d_out
  gemm_bf16_wmma<<<dim3(HID/128, SEQ/128), blk, 0, stream>>>(att, Wo, out, SEQ, HID, HID);
}
